// GCN_50053548868062
// MI455X (gfx1250) — matrix operand order repaired and checked by reading, not compiled, not run on hardware
//
#include <hip/hip_runtime.h>
#include <hip/hip_bf16.h>

#define N_NODES 100000
#define N_EDGES 6400000
#define IN_DIM  128
#define HID     16
#define CLASSES 40

typedef __attribute__((ext_vector_type(16))) _Float16 v16h;
typedef __attribute__((ext_vector_type(8)))  float    v8f;

// ---------------------------------------------------------------------------
// init: deg starts at 1.0 (self-loop weight); zero both scatter accumulators;
// build zero-padded W2 (32x16, rows 16..31 = 0) for the K=16 WMMA.
// ---------------------------------------------------------------------------
__global__ void k_init(float* __restrict__ deg, float* __restrict__ out1,
                       float* __restrict__ out2, const float* __restrict__ W2,
                       float* __restrict__ w2pad) {
    int i = blockIdx.x * blockDim.x + threadIdx.x;
    if (i < N_NODES * HID) { out1[i] = 0.0f; out2[i] = 0.0f; }
    if (i < N_NODES) deg[i] = 1.0f;
    if (i < 32 * 16) w2pad[i] = (i < 16 * 16) ? W2[i] : 0.0f;
}

// deg[col[e]] += w[e]; edge stream is single-use -> non-temporal loads
__global__ void k_degree(const long long* __restrict__ col,
                         const float* __restrict__ w, float* __restrict__ deg) {
    int e = blockIdx.x * blockDim.x + threadIdx.x;
    if (e >= N_EDGES) return;
    int c = (int)__builtin_nontemporal_load(&col[e]);
    float we = __builtin_nontemporal_load(&w[e]);
    unsafeAtomicAdd(&deg[c], we);
}

// in-place: deg -> dinv = rsqrt(deg)
__global__ void k_dinv(float* __restrict__ deg) {
    int n = blockIdx.x * blockDim.x + threadIdx.x;
    if (n < N_NODES) { float d = deg[n]; deg[n] = (d > 0.0f) ? rsqrtf(d) : 0.0f; }
}

// ---------------------------------------------------------------------------
// Fragment layouts per CDNA5 ISA 7.12.2 (wave32), f16 16x16x32:
//   A: lane = half*16 + m; elem i -> K = (i>>3)*16 + half*8 + (i&7)
//      => each 8-elem group g is 8 *contiguous* K values at g*16 + half*8
//   B: lane = half*16 + n; elem i -> K = (i>>3)*16 + half*8 + (i&7) (same as A)
//   C: lane = half*16 + n; vgpr j -> M = j + 8*half
// ---------------------------------------------------------------------------

// C[M x 16] = A[M x 128] @ B[128 x 16]; fully unrolled: 4x v_wmma per wave.
__global__ void __launch_bounds__(256) k_gemm128(const float* __restrict__ A,
                                                 const float* __restrict__ B,
                                                 float* __restrict__ C, int M) {
    int wave = (blockIdx.x * blockDim.x + threadIdx.x) >> 5;
    int lane = threadIdx.x & 31;
    if (wave >= (M >> 4)) return;        // wave-uniform: EXEC all-1 for WMMA
    int half = lane >> 4;
    int nm   = lane & 15;

    const float4* arow = (const float4*)(A + (size_t)(wave * 16 + nm) * 128);
    v8f acc = {};
#pragma unroll
    for (int kb = 0; kb < 128; kb += 32) {
        v16h a, b;
#pragma unroll
        for (int g = 0; g < 2; ++g) {
            int base = (kb >> 2) + g * 4 + half * 2;   // float4 index
            float4 p0 = arow[base];
            float4 p1 = arow[base + 1];
            a[g * 8 + 0] = (_Float16)p0.x;  a[g * 8 + 1] = (_Float16)p0.y;
            a[g * 8 + 2] = (_Float16)p0.z;  a[g * 8 + 3] = (_Float16)p0.w;
            a[g * 8 + 4] = (_Float16)p1.x;  a[g * 8 + 5] = (_Float16)p1.y;
            a[g * 8 + 6] = (_Float16)p1.z;  a[g * 8 + 7] = (_Float16)p1.w;
        }
#pragma unroll
        for (int i = 0; i < 16; ++i)
            b[i] = (_Float16)B[(kb + (i >> 3) * 16 + half * 8 + (i & 7)) * 16 + nm];
        acc = __builtin_amdgcn_wmma_f32_16x16x32_f16(false, a, false, b,
                                                     (short)0, acc, false, false);
    }
#pragma unroll
    for (int j = 0; j < 8; ++j)
        C[(size_t)(wave * 16 + j + half * 8) * 16 + nm] = acc[j];
}

// C[M x 16] = A[M x 16] @ W2pad[32 x 16]; K padded to 32, one v_wmma per wave.
// A pad is static: elems 0..7 <- K = half*8 + j (always valid), elems 8..15 = 0.
__global__ void __launch_bounds__(256) k_gemm16(const float* __restrict__ A,
                                                const float* __restrict__ Bpad,
                                                float* __restrict__ C, int M) {
    int wave = (blockIdx.x * blockDim.x + threadIdx.x) >> 5;
    int lane = threadIdx.x & 31;
    if (wave >= (M >> 4)) return;
    int half = lane >> 4;
    int nm   = lane & 15;

    const float4* arow = (const float4*)(A + (size_t)(wave * 16 + nm) * 16);
    float4 p0 = arow[half * 2];
    float4 p1 = arow[half * 2 + 1];

    v16h a = {};
    a[0] = (_Float16)p0.x;  a[1] = (_Float16)p0.y;
    a[2] = (_Float16)p0.z;  a[3] = (_Float16)p0.w;
    a[4] = (_Float16)p1.x;  a[5] = (_Float16)p1.y;
    a[6] = (_Float16)p1.z;  a[7] = (_Float16)p1.w;

    v16h b;
#pragma unroll
    for (int i = 0; i < 16; ++i)
        b[i] = (_Float16)Bpad[((i >> 3) * 16 + half * 8 + (i & 7)) * 16 + nm];

    v8f acc = {};
    acc = __builtin_amdgcn_wmma_f32_16x16x32_f16(false, a, false, b,
                                                 (short)0, acc, false, false);
#pragma unroll
    for (int j = 0; j < 8; ++j)
        C[(size_t)(wave * 16 + j + half * 8) * 16 + nm] = acc[j];
}

// ---------------------------------------------------------------------------
// Edge scatter: out[c][d] += dinv[r]*w*dinv[c] * h[r][d].
// 16 lanes share one edge; edge stream non-temporal; atomics resolve in L2
// (node accumulators are 6.4 MB << 192 MB L2).
// ---------------------------------------------------------------------------
__global__ void k_scatter(const long long* __restrict__ row,
                          const long long* __restrict__ col,
                          const float* __restrict__ w,
                          const float* __restrict__ dinv,
                          const float* __restrict__ h,
                          float* __restrict__ out) {
    int t = blockIdx.x * blockDim.x + threadIdx.x;
    int e = t >> 4;
    if (e >= N_EDGES) return;
    int d = t & 15;
    int r = (int)__builtin_nontemporal_load(&row[e]);
    int c = (int)__builtin_nontemporal_load(&col[e]);
    float we = __builtin_nontemporal_load(&w[e]);
    float norm = dinv[r] * we * dinv[c];
    unsafeAtomicAdd(&out[(size_t)c * HID + d], norm * h[(size_t)r * HID + d]);
}

// self-loop contribution (dinv[n]^2 * h[n][d]) + bias (+ optional relu), in place
__global__ void k_fuse(const float* __restrict__ h, const float* __restrict__ dinv,
                       const float* __restrict__ bias, float* __restrict__ out,
                       int relu) {
    int i = blockIdx.x * blockDim.x + threadIdx.x;
    if (i >= N_NODES * HID) return;
    int n = i >> 4;
    int d = i & 15;
    float di = dinv[n];
    float v  = out[i] + di * di * h[i] + bias[d];
    out[i] = relu ? fmaxf(v, 0.0f) : v;
}

// ---------------------------------------------------------------------------
// Fused classifier + log_softmax: one node per thread, 40 logits in registers.
// ---------------------------------------------------------------------------
__global__ void k_classify(const float* __restrict__ x, const float* __restrict__ Wc,
                           const float* __restrict__ bc, float* __restrict__ out) {
    int n = blockIdx.x * blockDim.x + threadIdx.x;
    if (n >= N_NODES) return;
    float xr[HID];
#pragma unroll
    for (int d = 0; d < HID; ++d) xr[d] = x[(size_t)n * HID + d];

    float lg[CLASSES];
    float m = -3.402823466e38f;
#pragma unroll
    for (int j = 0; j < CLASSES; ++j) {
        float acc = bc[j];
#pragma unroll
        for (int d = 0; d < HID; ++d) acc = fmaf(xr[d], Wc[d * CLASSES + j], acc);
        lg[j] = acc;
        m = fmaxf(m, acc);
    }
    float s = 0.0f;
#pragma unroll
    for (int j = 0; j < CLASSES; ++j) s += expf(lg[j] - m);
    float lse = logf(s);
#pragma unroll
    for (int j = 0; j < CLASSES; ++j)
        out[(size_t)n * CLASSES + j] = lg[j] - m - lse;
}

// ---------------------------------------------------------------------------
extern "C" void kernel_launch(void* const* d_in, const int* in_sizes, int n_in,
                              void* d_out, int out_size, void* d_ws, size_t ws_size,
                              hipStream_t stream) {
    (void)in_sizes; (void)n_in; (void)out_size; (void)ws_size;

    const float*     z  = (const float*)d_in[0];
    const long long* ei = (const long long*)d_in[1];   // int64 [2][E]
    const float*     ea = (const float*)d_in[2];
    const float*     W1 = (const float*)d_in[3];
    const float*     b1 = (const float*)d_in[4];
    const float*     W2 = (const float*)d_in[5];
    const float*     b2 = (const float*)d_in[6];
    const float*     Wc = (const float*)d_in[7];
    const float*     bc = (const float*)d_in[8];
    float*           out = (float*)d_out;

    const long long* row = ei;
    const long long* col = ei + N_EDGES;

    // workspace layout (floats): deg | h1 | out1 | h2 | out2 | w2pad (~26 MB)
    float* ws   = (float*)d_ws;
    float* deg  = ws;                      // N
    float* h1   = deg + N_NODES;           // N*16
    float* o1   = h1  + N_NODES * HID;     // N*16
    float* h2   = o1  + N_NODES * HID;     // N*16
    float* o2   = h2  + N_NODES * HID;     // N*16
    float* w2p  = o2  + N_NODES * HID;     // 32*16

    const int B   = 256;
    const int NH  = N_NODES * HID;                 // 1.6M
    const int ET  = N_EDGES * HID;                 // 102.4M (edge,dim) threads
    const int TLS = N_NODES / 16;                  // 6250 wmma tiles
    const int WPB = B / 32;                        // 8 waves per block

    k_init  <<<(NH + B - 1) / B, B, 0, stream>>>(deg, o1, o2, W2, w2p);
    k_degree<<<(N_EDGES + B - 1) / B, B, 0, stream>>>(col, ea, deg);
    k_dinv  <<<(N_NODES + B - 1) / B, B, 0, stream>>>(deg);

    // conv1: h1 = z @ W1 ; scatter ; x1 = relu(out1 + dinv^2*h1 + b1)
    k_gemm128<<<(TLS + WPB - 1) / WPB, B, 0, stream>>>(z, W1, h1, N_NODES);
    k_scatter<<<(ET + B - 1) / B, B, 0, stream>>>(row, col, ea, deg, h1, o1);
    k_fuse   <<<(NH + B - 1) / B, B, 0, stream>>>(h1, deg, b1, o1, 1);

    // conv2: h2 = x1 @ W2 ; scatter ; x2 = out2 + dinv^2*h2 + b2
    k_gemm16 <<<(TLS + WPB - 1) / WPB, B, 0, stream>>>(o1, w2p, h2, N_NODES);
    k_scatter<<<(ET + B - 1) / B, B, 0, stream>>>(row, col, ea, deg, h2, o2);
    k_fuse   <<<(NH + B - 1) / B, B, 0, stream>>>(h2, deg, b2, o2, 0);

    // classifier + log_softmax
    k_classify<<<(N_NODES + B - 1) / B, B, 0, stream>>>(o2, Wc, bc, out);
}
